// SelfInteractionUNet_81466939670708
// MI455X (gfx1250) — matrix operand order repaired and checked by reading, not compiled, not run on hardware
//
#include <hip/hip_runtime.h>
#include <stdint.h>

// CDNA5 (gfx1250) wave32 WMMA types
typedef __bf16 v16bf __attribute__((ext_vector_type(16)));
typedef float  v8f   __attribute__((ext_vector_type(8)));

namespace {
constexpr int   kM       = 256;   // channels
constexpr int   kRows    = 16;    // rows per workgroup = one WMMA M tile
constexpr int   kThreads = 256;   // 8 wave32s
constexpr float kSqrt2    = 1.41421356237309515f;
constexpr float kInvSqrt3 = 0.57735026918962584f;
constexpr float kEps      = 1e-6f;
constexpr float kInvS768  = 0.036084391824351615f;  // 1/sqrt(3M)
constexpr float kInvS512  = 0.044194173824159216f;  // 1/sqrt(2M)
constexpr float kInv16    = 0.0625f;                // 1/sqrt(M)
}  // namespace

__device__ __forceinline__ float siluf(float x) { return x / (1.0f + __expf(-x)); }

__device__ __forceinline__ v8f wmma_bf16(v16bf a, v16bf b, v8f c) {
  // (neg_a, A, neg_b, B, c_mod, C, reuse_a, reuse_b)
  return __builtin_amdgcn_wmma_f32_16x16x32_bf16(false, a, false, b, (short)0, c,
                                                 false, false);
}

// Stage one 32(K) x 256(col) fp32 weight slice from global into LDS as bf16,
// stored column-major sW[col*32 + k] so B fragments are contiguous 16B reads.
__device__ __forceinline__ void stage_w(__bf16* sW, const float* __restrict__ w,
                                        int stride, int tid) {
  const int k  = tid >> 3;          // 0..31
  const int c0 = (tid & 7) * 32;    // 0..224
  const float* src = w + (size_t)k * stride + c0;
#pragma unroll
  for (int i = 0; i < 32; ++i) sW[(c0 + i) * 32 + k] = (__bf16)src[i];
}

// B fragment, 16-bit 32x16: lanes 0-15 hold K=0..7,16..23 of col n, lanes 16-31 K=8..15,24..31.
__device__ __forceinline__ v16bf load_bfrag(const __bf16* sW, int ct, int lane) {
  const int n  = lane & 15;
  const int kb = (lane >> 4) << 3;  // 0 or 8
  const __bf16* p = sW + (ct * 16 + n) * 32 + kb;
  v16bf b;
#pragma unroll
  for (int i = 0; i < 16; ++i) b[i] = p[(i < 8) ? i : (i + 8)];  // -> 2x ds_load_b128
  return b;
}

// A fragment, 16-bit 16x32 layout: lane<16: M=lane, VGPR0-3 K=0..7, VGPR4-7 K=16..23;
// lane>=16: M=lane-16, K=8..15 and K=24..31. Virtual A row built on the fly:
//  mode 0: s   mode 1: s^2   mode 2: sTmp (ts0b / s2 / h)   mode 3: v_d   mode 4: sqrt2*s*v_d
__device__ __forceinline__ v16bf make_afrag(int mode, int d, int cbase, int lane,
                                            const float* sS, const float* sV,
                                            const float* sT) {
  const int m  = lane & 15;
  const int kb = (lane >> 4) * 8;  // 0 or 8
  const float* srow = sS + m * kM;
  const float* vrow = sV + d * (kRows * kM) + m * kM;
  const float* trow = sT + m * kM;
  v16bf a;
#pragma unroll
  for (int j = 0; j < 8; ++j) {
    const int kk = (j < 4) ? (kb + 2 * j) : (16 + kb + 2 * (j - 4));
    const int c  = cbase + kk;
    float x0, x1;
    if (mode == 0)      { x0 = srow[c]; x1 = srow[c + 1]; }
    else if (mode == 1) { float a0 = srow[c], a1 = srow[c + 1]; x0 = a0 * a0; x1 = a1 * a1; }
    else if (mode == 2) { x0 = trow[c]; x1 = trow[c + 1]; }
    else if (mode == 3) { x0 = vrow[c]; x1 = vrow[c + 1]; }
    else { x0 = kSqrt2 * srow[c] * vrow[c]; x1 = kSqrt2 * srow[c + 1] * vrow[c + 1]; }
    a[2 * j]     = (__bf16)x0;
    a[2 * j + 1] = (__bf16)x1;
  }
  return a;
}

// f32 C/D tile: VGPR i holds row i (lanes 0-15) / row i+8 (lanes 16-31), col=lane&15.
__device__ __forceinline__ void store_tile(float* dst, v8f acc, int ct, int lane) {
  const int r0  = (lane >> 4) * 8;
  const int col = ct * 16 + (lane & 15);
#pragma unroll
  for (int i = 0; i < 8; ++i) dst[(r0 + i) * kM + col] = acc[i];
}

__global__ __launch_bounds__(kThreads, 1)
void unet_wmma_kernel(const float* __restrict__ scalars,
                      const float* __restrict__ vectors,
                      const float* __restrict__ W0,
                      const float* __restrict__ W1,
                      const float* __restrict__ Wa,
                      const float* __restrict__ Wb,
                      float* __restrict__ out) {
  __shared__ __attribute__((aligned(16))) float  sS[kRows * kM];          // 16 KB
  __shared__ __attribute__((aligned(16))) float  sV[3 * kRows * kM];      // 48 KB [d][r][c]
  __shared__ __attribute__((aligned(16))) float  sTmp[kRows * kM];        // 16 KB
  __shared__ __attribute__((aligned(16))) __bf16 sW[2][kM * 32];          // 32 KB (double buffer)
  __shared__ __attribute__((aligned(16))) __bf16 skS[4][kRows * kM];      // 32 KB
  __shared__ __attribute__((aligned(16))) __bf16 skV[4][3 * kRows * kM];  // 96 KB
  __shared__ float sStat[3 * kRows];

  const int tid  = threadIdx.x;
  const int lane = tid & 31;
  const int wave = tid >> 5;
  const int n0   = blockIdx.x * kRows;

  // ---- load activations entirely via async LDS-DMA ----
  {
    const float* gs = scalars + (size_t)n0 * kM;
    const unsigned ldsS = (unsigned)(uintptr_t)(&sS[0]);
#pragma unroll
    for (int i = 0; i < 4; ++i) {
      const unsigned off = (unsigned)(tid + i * kThreads) * 16u;
      asm volatile("global_load_async_to_lds_b128 %0, %1, off"
                   :: "v"(ldsS + off), "v"((const void*)((const char*)gs + off))
                   : "memory");
    }
    // v: per-lane gathered DMA performs the [r][c][d] -> [d][r][c] transpose.
    const float* gv = vectors + (size_t)n0 * (3 * kM);
    const unsigned ldsV = (unsigned)(uintptr_t)(&sV[0]);
    for (int i = 0; i < 48; ++i) {
      const int idx = tid + i * kThreads;  // = d*4096 + r*256 + c
      const int d = idx >> 12, r = (idx >> 8) & 15, c = idx & 255;
      const float* ga = gv + (size_t)r * 768 + c * 3 + d;
      asm volatile("global_load_async_to_lds_b32 %0, %1, off"
                   :: "v"(ldsV + (unsigned)idx * 4u), "v"((const void*)ga)
                   : "memory");
    }
    asm volatile("s_wait_asynccnt 0" ::: "memory");
  }
  __syncthreads();

  for (int layer = 0; layer < 9; ++layer) {
    const float* W0l = W0 + (size_t)layer * 768 * 256;
    const float* W1l = W1 + (size_t)layer * 512 * 256;
    const float* Wal = Wa + (size_t)layer * 256 * 256;
    const float* Wbl = Wb + (size_t)layer * 256 * 512;

    if (layer >= 5) {  // decoder: add skip
      const int si = 8 - layer;
      for (int i = 0; i < 16; ++i) { int x = tid + i * 256; sS[x] += (float)skS[si][x]; }
      for (int i = 0; i < 48; ++i) { int x = tid + i * 256; sV[x] += (float)skV[si][x]; }
      __syncthreads();
    }

    // prologue-stage W0 slice 0 while computing ts0b = |v|^2/sqrt(3) -> sTmp
    stage_w(sW[0], W0l, 256, tid);
    for (int i = 0; i < 16; ++i) {
      const int idx = tid + i * 256, r = idx >> 8, c = idx & 255;
      const float a0 = sV[r * 256 + c], a1 = sV[4096 + r * 256 + c],
                  a2 = sV[8192 + r * 256 + c];
      sTmp[idx] = (a0 * a0 + a1 * a1 + a2 * a2) * kInvSqrt3;
    }
    __syncthreads();

    // ---- main GEMMs, double-buffered staging: one barrier per k-step ----
    v8f accS[2] = {};
    v8f accV[3][2] = {};
    for (int ks = 0; ks < 24; ++ks) {
      __builtin_prefetch(W0l + (size_t)(ks + 2) * 8192 + (size_t)tid * 32, 0, 3);
      if (ks + 1 < 24) stage_w(sW[(ks + 1) & 1], W0l + (size_t)(ks + 1) * 8192, 256, tid);
      const __bf16* wb = sW[ks & 1];
      const int mode = (ks < 8) ? 0 : (ks < 16) ? 1 : 2;
      v16bf a  = make_afrag(mode, 0, (ks & 7) * 32, lane, sS, sV, sTmp);
      v16bf b0 = load_bfrag(wb, 2 * wave, lane);
      v16bf b1 = load_bfrag(wb, 2 * wave + 1, lane);
      accS[0] = wmma_bf16(a, b0, accS[0]);
      accS[1] = wmma_bf16(a, b1, accS[1]);
      __syncthreads();
    }
    stage_w(sW[0], W1l, 256, tid);
    __syncthreads();
    for (int ks = 0; ks < 16; ++ks) {
      __builtin_prefetch(W1l + (size_t)(ks + 2) * 8192 + (size_t)tid * 32, 0, 3);
      if (ks + 1 < 16) stage_w(sW[(ks + 1) & 1], W1l + (size_t)(ks + 1) * 8192, 256, tid);
      const __bf16* wb = sW[ks & 1];
      const int mode = (ks < 8) ? 3 : 4;
      const int cb = (ks & 7) * 32;
      v16bf b0 = load_bfrag(wb, 2 * wave, lane);
      v16bf b1 = load_bfrag(wb, 2 * wave + 1, lane);
#pragma unroll
      for (int d = 0; d < 3; ++d) {
        v16bf a = make_afrag(mode, d, cb, lane, sS, sV, sTmp);
        accV[d][0] = wmma_bf16(a, b0, accV[d][0]);
        accV[d][1] = wmma_bf16(a, b1, accV[d][1]);
      }
      __syncthreads();
    }

    // fan-in normalization
#pragma unroll
    for (int t = 0; t < 2; ++t)
#pragma unroll
      for (int i = 0; i < 8; ++i) accS[t][i] *= kInvS768;
#pragma unroll
    for (int d = 0; d < 3; ++d)
#pragma unroll
      for (int t = 0; t < 2; ++t)
#pragma unroll
        for (int i = 0; i < 8; ++i) accV[d][t][i] *= kInvS512;

    // s2 -> sTmp as the MLP A-input (ts0b is dead; s2 also stays in regs)
    store_tile(sTmp, accS[0], 2 * wave, lane);
    store_tile(sTmp, accS[1], 2 * wave + 1, lane);
    stage_w(sW[0], Wal, 256, tid);
    __syncthreads();

    // h = silu(s2 @ Wa / 16)
    v8f accH[2] = {};
    for (int ks = 0; ks < 8; ++ks) {
      if (ks + 1 < 8) stage_w(sW[(ks + 1) & 1], Wal + (size_t)(ks + 1) * 8192, 256, tid);
      const __bf16* wb = sW[ks & 1];
      v16bf a  = make_afrag(2, 0, ks * 32, lane, sS, sV, sTmp);
      v16bf b0 = load_bfrag(wb, 2 * wave, lane);
      v16bf b1 = load_bfrag(wb, 2 * wave + 1, lane);
      accH[0] = wmma_bf16(a, b0, accH[0]);
      accH[1] = wmma_bf16(a, b1, accH[1]);
      __syncthreads();
    }
#pragma unroll
    for (int t = 0; t < 2; ++t)
#pragma unroll
      for (int i = 0; i < 8; ++i) accH[t][i] = siluf(accH[t][i] * kInv16);
    store_tile(sTmp, accH[0], 2 * wave, lane);   // h overwrites s2 copy
    store_tile(sTmp, accH[1], 2 * wave + 1, lane);
    stage_w(sW[0], Wbl, 512, tid);
    __syncthreads();

    // g = silu(h @ Wb / 16): wave-local g tiles align with this wave's s2/v2
    // tiles, so gating happens entirely in registers (g never materializes).
    v8f accG[2] = {};
    for (int ks = 0; ks < 8; ++ks) {
      if (ks + 1 < 8) stage_w(sW[(ks + 1) & 1], Wbl + (size_t)(ks + 1) * 16384, 512, tid);
      const __bf16* wb = sW[ks & 1];
      v16bf a  = make_afrag(2, 0, ks * 32, lane, sS, sV, sTmp);
      v16bf b0 = load_bfrag(wb, 2 * wave, lane);
      v16bf b1 = load_bfrag(wb, 2 * wave + 1, lane);
      accG[0] = wmma_bf16(a, b0, accG[0]);
      accG[1] = wmma_bf16(a, b1, accG[1]);
      __syncthreads();
    }
#pragma unroll
    for (int t = 0; t < 2; ++t)
#pragma unroll
      for (int i = 0; i < 8; ++i) accS[t][i] *= siluf(accG[t][i] * kInv16);

    v8f accG2[2] = {};
    stage_w(sW[0], Wbl + 256, 512, tid);
    __syncthreads();
    for (int ks = 0; ks < 8; ++ks) {
      if (ks + 1 < 8)
        stage_w(sW[(ks + 1) & 1], Wbl + (size_t)(ks + 1) * 16384 + 256, 512, tid);
      const __bf16* wb = sW[ks & 1];
      v16bf a  = make_afrag(2, 0, ks * 32, lane, sS, sV, sTmp);
      v16bf b0 = load_bfrag(wb, 2 * wave, lane);
      v16bf b1 = load_bfrag(wb, 2 * wave + 1, lane);
      accG2[0] = wmma_bf16(a, b0, accG2[0]);
      accG2[1] = wmma_bf16(a, b1, accG2[1]);
      __syncthreads();
    }
#pragma unroll
    for (int t = 0; t < 2; ++t)
#pragma unroll
      for (int i = 0; i < 8; ++i) {
        const float g2 = siluf(accG2[t][i] * kInv16);
#pragma unroll
        for (int d = 0; d < 3; ++d) accV[d][t][i] *= g2;
      }

    // gated outputs become the new (s, v)
    store_tile(sS, accS[0], 2 * wave, lane);
    store_tile(sS, accS[1], 2 * wave + 1, lane);
#pragma unroll
    for (int d = 0; d < 3; ++d) {
      store_tile(sV + d * 4096, accV[d][0], 2 * wave, lane);
      store_tile(sV + d * 4096, accV[d][1], 2 * wave + 1, lane);
    }
    __syncthreads();

    // equivariant layernorm (encoder 0-3 and decoder 5-7 only)
    const bool do_norm = (layer < 4) || (layer >= 5 && layer < 8);
    if (do_norm) {
      if (tid < 48) sStat[tid] = 0.0f;
      __syncthreads();
      const int r = tid >> 4, sub = tid & 15;
      float sum = 0.f, sq = 0.f;
      for (int i = 0; i < 16; ++i) {
        const float x = sS[r * 256 + sub * 16 + i];
        sum += x; sq += x * x;
      }
      atomicAdd(&sStat[r], sum);
      atomicAdd(&sStat[16 + r], sq);
      float vs = 0.f;
      for (int i = 0; i < 48; ++i) {
        const int e = sub * 48 + i;
        const float x = sV[(e >> 8) * 4096 + r * 256 + (e & 255)];
        vs += x * x;
      }
      atomicAdd(&sStat[32 + r], vs);
      __syncthreads();
      const float mu  = sStat[r] * (1.0f / 256.0f);
      const float var = sStat[16 + r] * (1.0f / 256.0f) - mu * mu;
      const float isg = rsqrtf(var + kEps);
      const float ivv = rsqrtf(sStat[32 + r] * (1.0f / 768.0f) + kEps);
      for (int i = 0; i < 16; ++i) {
        const int idx = r * 256 + sub * 16 + i;
        sS[idx] = (sS[idx] - mu) * isg;
      }
      for (int i = 0; i < 48; ++i) {
        const int e = sub * 48 + i;
        const int idx = (e >> 8) * 4096 + r * 256 + (e & 255);
        sV[idx] *= ivv;
      }
      __syncthreads();
    }

    if (layer < 4) {  // encoder: save skip (bf16)
      for (int i = 0; i < 16; ++i) { int x = tid + i * 256; skS[layer][x] = (__bf16)sS[x]; }
      for (int i = 0; i < 48; ++i) { int x = tid + i * 256; skV[layer][x] = (__bf16)sV[x]; }
      __syncthreads();
    }
  }

  // ---- output: [N, 4M] = [s | v.reshape(-1)] ----
  // scalar half: rows are contiguous in LDS and in global -> async store DMA
  {
    const unsigned ldsS = (unsigned)(uintptr_t)(&sS[0]);
    const char* gout = (const char*)(out + (size_t)n0 * 1024);
#pragma unroll
    for (int i = 0; i < 4; ++i) {
      const unsigned loff = (unsigned)(tid + i * kThreads) * 16u;  // byte off in sS
      const char* ga = gout + (size_t)(loff >> 10) * 4096 + (loff & 1023u);
      asm volatile("global_store_async_from_lds_b128 %0, %1, off"
                   :: "v"((const void*)ga), "v"(ldsS + loff) : "memory");
    }
  }
  for (int i = 0; i < 48; ++i) {
    const int idx = tid + i * 256;
    const int d = idx >> 12, r = (idx >> 8) & 15, c = idx & 255;
    out[(size_t)(n0 + r) * 1024 + 256 + c * 3 + d] = sV[idx];
  }
  asm volatile("s_wait_asynccnt 0" ::: "memory");
}

extern "C" void kernel_launch(void* const* d_in, const int* in_sizes, int n_in,
                              void* d_out, int out_size, void* d_ws, size_t ws_size,
                              hipStream_t stream) {
  (void)n_in; (void)out_size; (void)d_ws; (void)ws_size;
  const float* scalars = (const float*)d_in[0];
  const float* vectors = (const float*)d_in[1];
  const float* W0 = (const float*)d_in[2];
  const float* W1 = (const float*)d_in[3];
  const float* Wa = (const float*)d_in[4];
  const float* Wb = (const float*)d_in[5];
  float* out = (float*)d_out;

  const int N = in_sizes[0] / kM;          // 16384
  const int grid = N / kRows;              // 1024 workgroups of 16 rows
  hipLaunchKernelGGL(unet_wmma_kernel, dim3(grid), dim3(kThreads), 0, stream,
                     scalars, vectors, W0, W1, Wa, Wb, out);
}
